// SBMENet_48344151884075
// MI455X (gfx1250) — matrix operand order repaired and checked by reading, not compiled, not run on hardware
//
#include <hip/hip_runtime.h>
#include <stdint.h>

// ---------------- types ----------------
typedef __attribute__((ext_vector_type(16))) __bf16       v16bf;
typedef __attribute__((ext_vector_type(8)))  float        v8f;
typedef __attribute__((ext_vector_type(4))) unsigned int  u32x4;
typedef __attribute__((ext_vector_type(8))) unsigned int  u32x8;

__device__ __forceinline__ float lrelu(float v) { return v >= 0.f ? v : 0.1f * v; }

__device__ __forceinline__ unsigned short f2bf(float f) {
    unsigned u = __float_as_uint(f);
    u = u + 0x7FFFu + ((u >> 16) & 1u);   // round-to-nearest-even
    return (unsigned short)(u >> 16);
}

// ---------------- elementwise kernels (all 32-bit indexing) ----------------
__global__ void k_castn(const float* __restrict__ s, unsigned short* __restrict__ d, int n) {
    int i = blockIdx.x * blockDim.x + threadIdx.x;
    if (i < n) d[i] = f2bf(s[i]);
}

// cast channel range [c0,c1) of an [N][Ct][H][W] buffer into its bf16 mirror
__global__ void k_cast_rng(const float* __restrict__ s, unsigned short* __restrict__ d,
                           int Ct, int c0, int c1, int HW, int N) {
    int per = (c1 - c0) * HW;
    int i = blockIdx.x * blockDim.x + threadIdx.x;
    if (i >= per * N) return;
    int img = i / per;
    int rem = i - img * per;
    int off = (img * Ct + c0) * HW + rem;
    d[off] = f2bf(s[off]);
}

__global__ void k_add(float* __restrict__ dst, const float* __restrict__ src, int n) {
    int i = blockIdx.x * blockDim.x + threadIdx.x;
    if (i < n) dst[i] += src[i];
}

__global__ void k_copych(const float* __restrict__ src, int sCt, int sCoff,
                         float* __restrict__ dst, int dCt, int dCoff,
                         int N, int C, int HW) {
    int tot = N * C * HW;
    int i = blockIdx.x * blockDim.x + threadIdx.x;
    if (i >= tot) return;
    int sp = i % HW; int t = i / HW;
    int c = t % C;   int img = t / C;
    dst[(img * dCt + dCoff + c) * HW + sp] = src[(img * sCt + sCoff + c) * HW + sp];
}

// ---------------- WMMA implicit-GEMM 3x3 conv ----------------
// D = A(MxK) x B(KxN) + C ; M = N*Ho*Wo pixels, N = Cout, K = Cin*9 (chunked 32 per tap).
// Weight tiles (32 ci x 32 co bf16) double-buffered in LDS via Tensor Data Mover prefetch;
// 8 waves share them (2 cout tiles per wave -> 2 WMMAs per A fetch). A fragments come from
// branch-free clamped global loads with uniform offsets, zeroed via AND masks. Pixel
// coordinates are derived from TWO uniform scalar divisions + carry increments (no per-lane
// division anywhere in this kernel).
__global__ __launch_bounds__(256)
void k_conv3x3_wmma(const unsigned short* __restrict__ inb,  // bf16 [N][inCt][Hi][Wi]
                    const unsigned short* __restrict__ wb,   // bf16 [Cout][Cin][3][3]
                    const float* __restrict__ bias,
                    float* __restrict__ out,                 // f32  [N][outCt][Ho][Wo]
                    int N, int Cin, int Cout, int Hi, int Wi, int Ho, int Wo,
                    int inCt, int inCoff, int outCt, int outCoff,
                    int stride, int pad, int dil, int relu) {
    __shared__ unsigned sB[2][512];   // 2 x (32 co rows x 32 ci bf16)

    const int lane = threadIdx.x & 31;
    const int wave = threadIdx.x >> 5;
    const int half = lane >> 4;
    const int l15  = lane & 15;
    const int M    = N * Ho * Wo;
    const int HoWo = Ho * Wo;

    // ---- uniform wave-base pixel decomposition (scalar divisions) ----
    const int bwave = blockIdx.x * 128 + wave * 16;
    const int i0   = bwave / HoWo;            // uniform
    const int rem0 = bwave - i0 * HoWo;
    const int y0   = rem0 / Wo;               // uniform
    const int x0   = rem0 - y0 * Wo;

    // A-row pixel for this lane: bwave + l15 (carry-correct, Wo >= 8 so <= 2 wraps)
    int px = x0 + l15, py = y0, pimg = i0;
    if (px >= Wo) { px -= Wo; ++py; }
    if (px >= Wo) { px -= Wo; ++py; }
    if (py >= Ho) { py -= Ho; ++pimg; }
    if (py >= Ho) { py -= Ho; ++pimg; }
    const bool mok = (bwave + l15) < M;

    const int co0 = blockIdx.y * 32;
    const int HiWi = Hi * Wi;
    const int nchunk = (Cin + 31) >> 5;
    const int nit = 9 * nchunk;
    const int laneK = half * 8 * HiWi;        // fold divergent half-K into pointer
    const unsigned lds0 = (unsigned)(unsigned long long)(uintptr_t)&sB[0][0];

    // ---- TDM issue helper: 2D tile [32 ci][32 co] into LDS buffer ----
    auto issue_tdm = [&](int tp, int c0, unsigned laddr) {
        unsigned long long ga = (unsigned long long)(uintptr_t)wb +
            2ull * ((unsigned long long)(unsigned)(co0 * Cin + c0) * 9ull + (unsigned)tp);
        const unsigned td0 = (unsigned)(Cin - c0);   // remaining ci -> OOB zero fill
        const unsigned td1 = (unsigned)(Cout - co0); // remaining co -> OOB zero fill
        const unsigned long long s1 = (unsigned long long)(unsigned)Cin * 9ull;
        u32x4 g0 = {};
        g0[0] = 1u;                                              // count=1
        g0[1] = laddr;                                           // LDS byte address
        g0[2] = (unsigned)ga;                                    // global addr lo
        g0[3] = (unsigned)((ga >> 32) & 0x01FFFFFFull) | (2u << 30);  // hi | type=2
        u32x8 g1 = {};
        g1[0] = (1u << 16);                                      // data_size=2B
        g1[1] = (td0 & 0xFFFFu) << 16;                           // tensor_dim0 lo16
        g1[2] = (td0 >> 16) | ((td1 & 0xFFFFu) << 16);           // dim0 hi | dim1 lo
        g1[3] = (td1 >> 16) | (32u << 16);                       // dim1 hi | tile_dim0=32
        g1[4] = 32u;                                             // tile_dim1=32 co rows
        g1[5] = 9u;                                              // dim0 stride (elems)
        g1[6] = (unsigned)((s1 & 0xFFFFull) << 16);              // dim1 stride lo16
        g1[7] = (unsigned)(s1 >> 16);                            // dim1 stride hi
        asm volatile("tensor_load_to_lds %0, %1" :: "s"(g0), "s"(g1) : "memory");
    };

    if (wave == 0) issue_tdm(0, 0, lds0);     // prefetch first tile

    v8f acc0 = {}, acc1 = {};
    int it = 0;
    for (int tap = 0; tap < 9; ++tap) {
        const int kh = tap / 3, kw = tap - kh * 3;
        const int yi = py * stride - pad + kh * dil;
        const int xi = px * stride - pad + kw * dil;
        const bool ok = mok && yi >= 0 && yi < Hi && xi >= 0 && xi < Wi;
        // clamped per-lane base (always safe to read [0, Cin*HiWi) elements from it)
        const unsigned short* aptr =
            (ok ? (inb + (pimg * inCt + inCoff) * HiWi + yi * Wi + xi) : inb) + laneK;
        const unsigned amask = ok ? 0xFFFFFFFFu : 0u;

        for (int cc = 0; cc < nchunk; ++cc, ++it) {
            const int ci0 = cc << 5;
            const int cur = it & 1;
            __syncthreads();                   // prev reads of buf[!cur] complete
            if (wave == 0) {
                if (it + 1 < nit) {            // prefetch next chunk's weights
                    int tn = tap, cn = ci0 + 32;
                    if (cc == nchunk - 1) { tn = tap + 1; cn = 0; }
                    issue_tdm(tn, cn, lds0 + (unsigned)((it + 1) & 1) * 2048u);
                    __builtin_amdgcn_s_wait_tensorcnt(1);   // current tile done
                } else {
                    __builtin_amdgcn_s_wait_tensorcnt(0);
                }
            }
            __syncthreads();

            // B fragments from LDS: lane col n=l15, VGPR j holds K={(j&4)*4+half*8+2*(j&3), +1}
            const unsigned* sp_ = &sB[cur][0];
            union { unsigned u[8]; v16bf v; } B0, B1;
            {
                const int b0 = l15 * 16 + half * 4;
                #pragma unroll
                for (int j = 0; j < 8; ++j) B0.u[j] = sp_[b0 + j + (j & 4)];
                #pragma unroll
                for (int j = 0; j < 8; ++j) B1.u[j] = sp_[b0 + 256 + j + (j & 4)];
            }

            // A fragment: 16 unconditional loads at uniform offsets, zero via AND mask
            union { unsigned u[8]; v16bf v; } Af;
            if (ci0 + 32 <= Cin) {             // steady-state fast path (uniform branch)
                #pragma unroll
                for (int j = 0; j < 8; ++j) {
                    const int basej = ((j & 4) << 2) + ((j & 3) << 1);   // 0,2,4,6,16,18,20,22
                    const int e = (ci0 + basej) * HiWi;                  // uniform
                    unsigned t0 = aptr[e];
                    unsigned t1 = aptr[e + HiWi];
                    Af.u[j] = (t0 | (t1 << 16)) & amask;
                }
            } else {                           // K tail: per-element masks, still branch-free
                #pragma unroll
                for (int j = 0; j < 8; ++j) {
                    const int basej = ((j & 4) << 2) + ((j & 3) << 1);
                    const int k0 = ci0 + basej + half * 8;
                    const unsigned m0 = (ok && k0 < Cin)     ? 0xFFFFu : 0u;
                    const unsigned m1 = (ok && k0 + 1 < Cin) ? 0xFFFFu : 0u;
                    const int e0 = (k0 < Cin)     ? (ci0 + basej) * HiWi : 0;
                    const int e1 = (k0 + 1 < Cin) ? (ci0 + basej) * HiWi + HiWi : 0;
                    unsigned t0 = aptr[e0] & m0;
                    unsigned t1 = aptr[e1] & m1;
                    Af.u[j] = t0 | (t1 << 16);
                }
            }

            acc0 = __builtin_amdgcn_wmma_f32_16x16x32_bf16(false, Af.v, false, B0.v,
                                                           (short)0, acc0, false, false);
            acc1 = __builtin_amdgcn_wmma_f32_16x16x32_bf16(false, Af.v, false, B1.v,
                                                           (short)0, acc1, false, false);
        }
    }

    // ---- store: C layout VGPR r -> M = r + half*8 (relative to wave base); N = l15 ----
    const int coA = co0 + l15, coB = co0 + 16 + l15;
    const float bA = (coA < Cout) ? bias[coA] : 0.f;
    const float bB = (coB < Cout) ? bias[coB] : 0.f;
    int sx = x0 + half * 8, sy = y0, si = i0;
    if (sx >= Wo) { sx -= Wo; ++sy; }
    if (sx >= Wo) { sx -= Wo; ++sy; }
    if (sy >= Ho) { sy -= Ho; ++si; }
    if (sy >= Ho) { sy -= Ho; ++si; }
    #pragma unroll
    for (int r = 0; r < 8; ++r) {
        const int mm = bwave + half * 8 + r;
        if (mm < M) {
            const int base = ((si * outCt + outCoff) * Ho + sy) * Wo + sx;
            if (coA < Cout) {
                float v = acc0[r] + bA; if (relu) v = lrelu(v);
                out[base + coA * HoWo] = v;
            }
            if (coB < Cout) {
                float v = acc1[r] + bB; if (relu) v = lrelu(v);
                out[base + coB * HoWo] = v;
            }
        }
        ++sx;
        if (sx == Wo) { sx = 0; ++sy; if (sy == Ho) { sy = 0; ++si; } }
    }
}

// ---------------- deconv 4x4 s2 p1 (torch ConvTranspose2d, w[Cin][Cout][4][4]) ----------------
__global__ void k_deconv4(const float* __restrict__ in, const float* __restrict__ w,
                          const float* __restrict__ bias, float* __restrict__ out,
                          int N, int Cin, int Cout, int Hi, int Wi) {
    const int Ho = Hi * 2, Wo = Wi * 2;
    int tot = N * Cout * Ho * Wo;
    int i = blockIdx.x * blockDim.x + threadIdx.x;
    if (i >= tot) return;
    int x = i % Wo; int t = i / Wo;
    int y = t % Ho; t /= Ho;
    int co = t % Cout; int img = t / Cout;
    const int istr = Hi * Wi;
    const int wstr = Cout * 16;
    float a = bias[co];
    for (int ky = 0; ky < 4; ++ky) {
        int yy = y + 1 - ky; if (yy & 1) continue;
        int iy = yy >> 1;    if (iy < 0 || iy >= Hi) continue;
        for (int kx = 0; kx < 4; ++kx) {
            int xx = x + 1 - kx; if (xx & 1) continue;
            int ix = xx >> 1;    if (ix < 0 || ix >= Wi) continue;
            const float* ip = in + img * Cin * istr + iy * Wi + ix;
            const float* wp = w + co * 16 + ky * 4 + kx;
            float s = 0.f;
            for (int ci = 0; ci < Cin; ++ci) s += ip[ci * istr] * wp[ci * wstr];
            a += s;
        }
    }
    out[((img * Cout + co) * Ho + y) * Wo + x] = a;
}

// ---------------- channel l2norm ----------------
__global__ void k_l2norm(const float* __restrict__ in, float* __restrict__ out,
                         int N, int C, int HW) {
    int tot = N * HW;
    int i = blockIdx.x * blockDim.x + threadIdx.x;
    if (i >= tot) return;
    int img = i / HW, sp = i % HW;
    const float* p = in + img * C * HW + sp;
    float s = 1e-6f;
    for (int c = 0; c < C; ++c) { float v = p[c * HW]; s += v * v; }
    float r = rsqrtf(s);
    float* o = out + img * C * HW + sp;
    for (int c = 0; c < C; ++c) o[c * HW] = p[c * HW] * r;
}

// ---------------- bilinear sampling helpers (reference _sample semantics) ----------------
struct Bil { int x0, y0, x1, y1; float w00, w10, w01, w11, keep; };
__device__ __forceinline__ Bil mkbil(float gx, float gy, int H, int W) {
    float fx = floorf(gx), fy = floorf(gy);
    float tx = gx - fx, ty = gy - fy;
    int x0 = (int)fx, y0 = (int)fy, x1 = x0 + 1, y1 = y0 + 1;
    float v00 = (x0 >= 0 && x0 < W && y0 >= 0 && y0 < H) ? 1.f : 0.f;
    float v10 = (x1 >= 0 && x1 < W && y0 >= 0 && y0 < H) ? 1.f : 0.f;
    float v01 = (x0 >= 0 && x0 < W && y1 >= 0 && y1 < H) ? 1.f : 0.f;
    float v11 = (x1 >= 0 && x1 < W && y1 >= 0 && y1 < H) ? 1.f : 0.f;
    float m = (v00 * (1 - tx) + v10 * tx) * (1 - ty) + (v01 * (1 - tx) + v11 * tx) * ty;
    Bil b;
    b.w00 = (1 - tx) * (1 - ty); b.w10 = tx * (1 - ty);
    b.w01 = (1 - tx) * ty;       b.w11 = tx * ty;
    b.x0 = min(max(x0, 0), W - 1); b.x1 = min(max(x1, 0), W - 1);
    b.y0 = min(max(y0, 0), H - 1); b.y1 = min(max(y1, 0), H - 1);
    b.keep = (m < 0.999f) ? 0.f : 1.f;
    return b;
}
__device__ __forceinline__ float bsamp(const float* ch, int W, const Bil& b) {
    return ch[b.y0 * W + b.x0] * b.w00 + ch[b.y0 * W + b.x1] * b.w10 +
           ch[b.y1 * W + b.x0] * b.w01 + ch[b.y1 * W + b.x1] * b.w11;
}

// out[:,c,y,x] = sample(src, (x,y) + scale*flow) * validity
__global__ void k_warp(const float* __restrict__ src, const float* __restrict__ flow,
                       float scale, float* __restrict__ out,
                       int N, int C, int H, int W, int outCt, int outCoff) {
    int HW = H * W;
    int tot = N * HW;
    int i = blockIdx.x * blockDim.x + threadIdx.x;
    if (i >= tot) return;
    int img = i / HW; int sp = i % HW;
    int y = sp / W, x = sp % W;
    float fx = flow[(img * 2 + 0) * HW + sp] * scale;
    float fy = flow[(img * 2 + 1) * HW + sp] * scale;
    Bil b = mkbil((float)x + fx, (float)y + fy, H, W);
    const float* sb = src + img * C * HW;
    float* ob = out + (img * outCt + outCoff) * HW + sp;
    for (int c = 0; c < C; ++c)
        ob[c * HW] = bsamp(sb + c * HW, W, b) * b.keep;
}

// bilateral correlation (t=0.5): out[n,r,y,x] = sum_c f1n(x-d) * f2n(x+d), d = scale*flow + disp
__global__ void k_corr(const float* __restrict__ f1n, const float* __restrict__ f2n,
                       const float* __restrict__ flow, float scale, int md,
                       float* __restrict__ out, int N, int C, int H, int W,
                       int outCt, int outCoff, int relu) {
    const int D = 2 * md + 1, R = D * D;
    int HW = H * W;
    int tot = N * R * HW;
    int i = blockIdx.x * blockDim.x + threadIdx.x;
    if (i >= tot) return;
    int sp = i % HW; int t = i / HW;
    int r = t % R;   int img = t / R;
    int y = sp / W, x = sp % W;
    float du = (float)(r % D - md), dv = (float)(r / D - md);
    float sx = 0.f, sy = 0.f;
    if (flow) {
        sx = flow[(img * 2 + 0) * HW + sp] * scale;
        sy = flow[(img * 2 + 1) * HW + sp] * scale;
    }
    float dx = sx + du, dy = sy + dv;
    Bil b1 = mkbil((float)x - dx, (float)y - dy, H, W);   // bw: warp f1 by -sbm_d
    Bil b2 = mkbil((float)x + dx, (float)y + dy, H, W);   // fw: warp f2 by +sbm_d
    const float* p1 = f1n + img * C * HW;
    const float* p2 = f2n + img * C * HW;
    float s = 0.f;
    for (int c = 0; c < C; ++c)
        s += bsamp(p1 + c * HW, W, b1) * bsamp(p2 + c * HW, W, b2);
    s *= b1.keep * b2.keep;
    if (relu) s = lrelu(s);
    out[(img * outCt + outCoff + r) * HW + sp] = s;
}

// ================= host orchestration =================
struct Arena { char* p; size_t off; };
static float* af(Arena& a, long long n) {
    a.off = (a.off + 255) & ~(size_t)255;
    float* r = (float*)(a.p + a.off); a.off += (size_t)n * 4; return r;
}
static unsigned short* ab(Arena& a, long long n) {
    a.off = (a.off + 255) & ~(size_t)255;
    unsigned short* r = (unsigned short*)(a.p + a.off); a.off += (size_t)n * 2; return r;
}
static inline unsigned gdiv(long long a, int b) { return (unsigned)((a + b - 1) / b); }

extern "C" void kernel_launch(void* const* d_in, const int* in_sizes, int n_in,
                              void* d_out, int out_size, void* d_ws, size_t ws_size,
                              hipStream_t stream) {
    (void)in_sizes; (void)n_in; (void)out_size; (void)ws_size;
    const int N = 4;
    const float* im1 = (const float*)d_in[0];
    const float* im2 = (const float*)d_in[1];

    // --- parameter walk (dict order of init_params) ---
    int q = 2;
    auto nxt = [&]() { return (const float*)d_in[q++]; };
    const float *ew[18], *ebias[18];
    for (int i = 0; i < 18; ++i) { ew[i] = nxt(); ebias[i] = nxt(); }
    const float *cw[5][5], *cb[5][5], *pfw[5], *pfb[5], *dvw[5], *dvb[5], *ufw[4], *ufb[4];
    for (int li = 0; li < 5; ++li) {                    // li: 0->lvl6 ... 4->lvl2
        for (int k = 0; k < 5; ++k) { cw[li][k] = nxt(); cb[li][k] = nxt(); }
        pfw[li] = nxt(); pfb[li] = nxt();
        dvw[li] = nxt(); dvb[li] = nxt();
        if (li < 4) { ufw[li] = nxt(); ufb[li] = nxt(); }
    }
    const float *dcw[7], *dcb[7];
    for (int i = 0; i < 7; ++i) { dcw[i] = nxt(); dcb[i] = nxt(); }

    Arena A{(char*)d_ws, 0};
    unsigned short* wscr = ab(A, 1200000);       // bf16 weight staging (max Cout*Cin*9)
    float* upflow = af(A, (long long)N * 2 * 128 * 128);
    float* upfeat = af(A, (long long)N * 2 * 128 * 128);

    const int fC[5] = {32, 64, 96, 128, 196};    // c2..c6 channels
    const int fH[5] = {128, 64, 32, 16, 8};
    float* cf[2][5]; unsigned short* cbm[2][5];
    for (int im = 0; im < 2; ++im)
        for (int l = 0; l < 5; ++l) {
            long long n = (long long)N * fC[l] * fH[l] * fH[l];
            cf[im][l] = af(A, n); cbm[im][l] = ab(A, n);
        }
    const size_t mark = A.off;

    auto cast = [&](const float* s, unsigned short* d, long long n) {
        k_castn<<<gdiv(n, 256), 256, 0, stream>>>(s, d, (int)n);
    };
    auto cast_rng = [&](const float* s, unsigned short* d, int Ct, int c0, int c1, int HW) {
        long long n = (long long)N * (c1 - c0) * HW;
        k_cast_rng<<<gdiv(n, 256), 256, 0, stream>>>(s, d, Ct, c0, c1, HW, N);
    };
    auto conv = [&](const unsigned short* inB, int inCt, int inCoff, int Cin, int Hi, int Wi,
                    const float* w, const float* b, int Cout, int stride, int pad, int dil,
                    int relu, float* outF, int outCt, int outCoff, unsigned short* outB) {
        cast(w, wscr, (long long)Cout * Cin * 9);
        int Ho = (Hi + 2 * pad - 2 * dil - 1) / stride + 1;
        int Wo = (Wi + 2 * pad - 2 * dil - 1) / stride + 1;
        long long M = (long long)N * Ho * Wo;
        dim3 g(gdiv(M, 128), (unsigned)((Cout + 31) / 32), 1);
        k_conv3x3_wmma<<<g, 256, 0, stream>>>(inB, wscr, b, outF, N, Cin, Cout, Hi, Wi, Ho, Wo,
                                              inCt, inCoff, outCt, outCoff, stride, pad, dil, relu);
        if (outB) cast_rng(outF, outB, outCt, outCoff, outCoff + Cout, Ho * Wo);
    };

    // --- feature pyramids ---
    auto pyramid = [&](const float* im, int idx) {
        size_t save = A.off;
        long long mx = (long long)N * 16 * 256 * 256;
        float* p0 = af(A, mx); float* p1 = af(A, mx);
        unsigned short* b0 = ab(A, mx); unsigned short* b1 = ab(A, mx);
        unsigned short* imB = ab(A, (long long)N * 3 * 512 * 512);
        cast(im, imB, (long long)N * 3 * 512 * 512);
        const unsigned short* cur = imB; int C = 3, H = 512;
        const int gC[6] = {16, 32, 64, 96, 128, 196};
        int wi = 0;
        for (int g = 0; g < 6; ++g) {
            int Co = gC[g];
            conv(cur, C, 0, C, H, H, ew[wi], ebias[wi], Co, 2, 1, 1, 1, p0, Co, 0, b0); ++wi;
            H /= 2;
            conv(b0, Co, 0, Co, H, H, ew[wi], ebias[wi], Co, 1, 1, 1, 1, p1, Co, 0, b1); ++wi;
            if (g == 0) {
                conv(b1, Co, 0, Co, H, H, ew[wi], ebias[wi], Co, 1, 1, 1, 1, p0, Co, 0, b0); ++wi;
                cur = b0; C = Co;
            } else {
                int l = g - 1;
                conv(b1, Co, 0, Co, H, H, ew[wi], ebias[wi], Co, 1, 1, 1, 1,
                     cf[idx][l], Co, 0, cbm[idx][l]); ++wi;
                cur = cbm[idx][l]; C = Co;
            }
        }
        A.off = save;
    };
    pyramid(im1, 0);
    pyramid(im2, 1);

    // --- coarse-to-fine refinement ---
    const int OD[5] = {169, 341, 277, 157, 93};
    const int MD[5] = {6, 4, 4, 2, 2};
    const float SCL[5] = {0.f, 0.625f, 1.25f, 2.5f, 5.0f};
    const long long FOFF[5] = {174080, 172032, 163840, 131072, 0};  // flow6..flow2
    const int OUTS[5] = {128, 128, 96, 64, 32};
    float* dout = (float*)d_out;

    for (int li = 0; li < 5; ++li) {
        const int fi = 4 - li;                       // feature index (c6..c2)
        const int H = fH[fi], C = fC[fi];
        const int T = OD[li] + 448;
        const int HW = H * H;
        A.off = mark;
        float* xF = af(A, (long long)N * T * HW);
        unsigned short* xB = ab(A, (long long)N * T * HW);
        float* f1n = af(A, (long long)N * C * HW);
        float* f2n = af(A, (long long)N * C * HW);

        long long nsp = (long long)N * HW;
        k_l2norm<<<gdiv(nsp, 256), 256, 0, stream>>>(cf[0][fi], f1n, N, C, HW);
        k_l2norm<<<gdiv(nsp, 256), 256, 0, stream>>>(cf[1][fi], f2n, N, C, HW);

        const int R = (2 * MD[li] + 1) * (2 * MD[li] + 1);
        const float* flw = (li == 0) ? nullptr : upflow;
        long long nc = (long long)N * R * HW;
        k_corr<<<gdiv(nc, 256), 256, 0, stream>>>(f1n, f2n, flw, SCL[li], MD[li],
                                                  xF, N, C, H, H, T, 448, 1);
        if (li > 0) {
            k_warp<<<gdiv(nsp, 256), 256, 0, stream>>>(cf[0][fi], upflow, -SCL[li],
                                                       xF, N, C, H, H, T, 448 + R);
            k_warp<<<gdiv(nsp, 256), 256, 0, stream>>>(cf[1][fi], upflow, SCL[li],
                                                       xF, N, C, H, H, T, 448 + R + C);
            long long n2 = (long long)N * 2 * HW;
            k_copych<<<gdiv(n2, 256), 256, 0, stream>>>(upflow, 2, 0, xF, T, 448 + R + 2 * C,
                                                        N, 2, HW);
            k_copych<<<gdiv(n2, 256), 256, 0, stream>>>(upfeat, 2, 0, xF, T, 448 + R + 2 * C + 2,
                                                        N, 2, HW);
        }
        cast_rng(xF, xB, T, 448, T, HW);

        // dense block (front-concat via channel offsets)
        int S = 448;
        for (int k = 0; k < 5; ++k) {
            conv(xB, T, S, T - S, H, H, cw[li][k], cb[li][k], OUTS[k], 1, 1, 1, 1,
                 xF, T, S - OUTS[k], xB);
            S -= OUTS[k];
        }
        // predict_flow -> directly into d_out slice
        conv(xB, T, 0, T, H, H, pfw[li], pfb[li], 2, 1, 1, 1, 0,
             dout + FOFF[li], 2, 0, nullptr);

        if (li < 4) {
            long long tf = (long long)N * 2 * (2 * H) * (2 * H);
            k_deconv4<<<gdiv(tf, 256), 256, 0, stream>>>(dout + FOFF[li], dvw[li], dvb[li],
                                                         upflow, N, 2, 2, H, H);
            k_deconv4<<<gdiv(tf, 256), 256, 0, stream>>>(xF, ufw[li], ufb[li],
                                                         upfeat, N, T, 2, H, H);
        } else {
            // context network on level-2 features
            long long yn = (long long)N * 128 * HW;
            float* y0 = af(A, yn); float* y1 = af(A, yn);
            unsigned short* yb0 = ab(A, yn); unsigned short* yb1 = ab(A, yn);
            float* ftmp = af(A, (long long)N * 2 * HW);
            const int cins[7]  = {T, 128, 128, 128, 96, 64, 32};
            const int couts[7] = {128, 128, 128, 96, 64, 32, 2};
            const int pads[7]  = {1, 2, 4, 8, 16, 1, 1};
            float* bufF[2] = {y0, y1}; unsigned short* bufB[2] = {yb0, yb1};
            const unsigned short* inB = xB; int ct = T;
            for (int i = 0; i < 7; ++i) {
                float* of = (i == 6) ? ftmp : bufF[i & 1];
                unsigned short* ob = (i == 6) ? nullptr : bufB[i & 1];
                conv(inB, ct, 0, cins[i], H, H, dcw[i], dcb[i], couts[i],
                     1, pads[i], pads[i], (i < 6) ? 1 : 0, of, couts[i], 0, ob);
                inB = bufB[i & 1]; ct = couts[i];
            }
            long long nf = (long long)N * 2 * HW;
            k_add<<<gdiv(nf, 256), 256, 0, stream>>>(dout, ftmp, (int)nf);   // flow2 += dc_conv7
        }
    }
}
